// VectorQuantizer_40355512714051
// MI455X (gfx1250) — compile-verified, matrix operand order checked by reading, not run on hardware
//
#include <hip/hip_runtime.h>
#include <hip/hip_bf16.h>

// ---------------------------------------------------------------------------
// VQ-VAE vector quantizer for MI455X (gfx1250, wave32, WMMA).
// Nearest-code search = 16384x8192x256 GEMM -> bf16x3-split WMMA
// (v_wmma_f32_16x16x32_bf16, f32 accumulate).
//  - 64 rows/block, 8 waves = 4 row-tiles x 2 column parities
//  - A-hi AND A-lo hoisted in VGPRs (loop-invariant, reused 256x)
//  - Weight tiles hi/lo in LDS (32KB), granule swizzle (g ^= row&15)
//    -> conflict-free ds_load_b128
//  - global_prefetch_b8 for next weight tile
// ---------------------------------------------------------------------------

#define NROWS   16384
#define DDIM    256
#define KCODES  8192
#define ROWS_PB 64
#define NJ      (KCODES / 32)   // 256 staging iterations of 32 codes

typedef __attribute__((ext_vector_type(16))) __bf16          v16bf;
typedef __attribute__((ext_vector_type(8)))  float           v8f;
typedef __attribute__((ext_vector_type(16))) unsigned short  us16;

__device__ __forceinline__ float bf2f(unsigned short h) {
  return __uint_as_float(((unsigned int)h) << 16);
}
// round-split v into bf16 hi + bf16 lo (returned as 16-bit patterns)
__device__ __forceinline__ void split2(float v, unsigned int& hb, unsigned int& lb) {
  unsigned int u = __float_as_uint(v);
  unsigned int r = u + 0x7FFFu + ((u >> 16) & 1u);
  hb = r >> 16;
  float hf = __uint_as_float(r & 0xFFFF0000u);
  float lo = v - hf;
  unsigned int u2 = __float_as_uint(lo);
  unsigned int r2 = u2 + 0x7FFFu + ((u2 >> 16) & 1u);
  lb = r2 >> 16;
}

// Load a WMMA operand: two non-adjacent 16B granules from LDS.
__device__ __forceinline__ v16bf ld2(const unsigned short* p0,
                                     const unsigned short* p1) {
  struct P { uint4 a, b; } t;
  t.a = *(const uint4*)p0;
  t.b = *(const uint4*)p1;
  return __builtin_bit_cast(v16bf, t);
}

// ------------------------- codebook squared norms --------------------------
__global__ __launch_bounds__(256) void wnorm_kernel(const float* __restrict__ w,
                                                    float* __restrict__ wnorm) {
  int wv   = (blockIdx.x * blockDim.x + threadIdx.x) >> 5;
  int lane = threadIdx.x & 31;
  if (wv >= KCODES) return;
  const float* row = w + (size_t)wv * DDIM;
  float s = 0.f;
  #pragma unroll
  for (int d = 0; d < DDIM / 32; ++d) {
    float v = row[lane + 32 * d];
    s += v * v;
  }
  #pragma unroll
  for (int m = 16; m >= 1; m >>= 1) s += __shfl_xor(s, m, 32);
  if (lane == 0) wnorm[wv] = s;
}

// ------------------------------ zero scratch -------------------------------
__global__ void zero_kernel(float* __restrict__ p, size_t n) {
  size_t stride = (size_t)gridDim.x * blockDim.x;
  for (size_t i = blockIdx.x * (size_t)blockDim.x + threadIdx.x; i < n; i += stride)
    p[i] = 0.f;
}

// --------------------- fused distance GEMM + argmin ------------------------
__global__ __launch_bounds__(256) void vq_argmin_kernel(
    const float* __restrict__ z, const float* __restrict__ w,
    const float* __restrict__ wnorm, int* __restrict__ idx_out,
    float* __restrict__ idx_f_out) {
  // 32KB LDS:
  //   whi32 : 32 codes x 128 uints (weight hi-split, swizzled) 16KB
  //   wlo32 : 32 codes x 128 uints (weight lo-split, swizzled) 16KB
  __shared__ unsigned int smem_u[8192];
  unsigned int* whi32 = smem_u;
  unsigned int* wlo32 = smem_u + 4096;
  const unsigned short* whi_us = (const unsigned short*)whi32;
  const unsigned short* wlo_us = (const unsigned short*)wlo32;

  const int tid  = threadIdx.x;
  const int lane = tid & 31;
  const int wv   = tid >> 5;
  const int rw   = wv & 3;     // row-tile 0..3 (16 rows each)
  const int c    = wv >> 2;    // column parity 0..1
  const int r16  = lane & 15;
  const int kh   = lane >> 4;  // K-half select per WMMA A/B layout
  const int row0 = blockIdx.x * ROWS_PB;

  // ---- hoist A hi AND lo (this wave's 16 z rows) into VGPRs.
  //      Both are invariant over the whole 256-iteration j loop.
  v16bf ahi[8], alo[8];
  {
    const float* zrow = z + (size_t)(row0 + rw * 16 + r16) * DDIM;
    #pragma unroll
    for (int kc = 0; kc < 8; ++kc) {
      const float* p = zrow + kc * 32 + kh * 8;
      us16 hh, ll;
      #pragma unroll
      for (int jj = 0; jj < 8; ++jj) {
        unsigned int h1, l1, h2, l2;
        split2(p[jj], h1, l1);
        split2(p[16 + jj], h2, l2);
        hh[jj]     = (unsigned short)h1;
        ll[jj]     = (unsigned short)l1;
        hh[8 + jj] = (unsigned short)h2;
        ll[8 + jj] = (unsigned short)l2;
      }
      ahi[kc] = __builtin_bit_cast(v16bf, hh);
      alo[kc] = __builtin_bit_cast(v16bf, ll);
    }
  }

  float best[8];
  int   bidx[8];
  #pragma unroll
  for (int r = 0; r < 8; ++r) { best[r] = 3.4e38f; bidx[r] = 0; }

  for (int j = 0; j < NJ; ++j) {
    __syncthreads();
    // ---- stage 32 codes [32j, 32j+32) into LDS, hi/lo split, swizzled ----
    const float* wt = w + (size_t)j * 32 * DDIM;
    #pragma unroll 4
    for (int i = 0; i < 16; ++i) {
      int p   = tid + 256 * i;      // 4096 pairs
      int row = p >> 7;
      int e   = (p << 1) & 255;
      float2 v2 = *(const float2*)(wt + 2 * p);
      unsigned int h1, l1, h2, l2;
      split2(v2.x, h1, l1);
      split2(v2.y, h2, l2);
      int gs    = (e >> 3) ^ (row & 15);
      int idx32 = row * 128 + gs * 4 + ((e >> 1) & 3);
      whi32[idx32] = h1 | (h2 << 16);
      wlo32[idx32] = l1 | (l2 << 16);
    }
    // prefetch next tile: one 128B cacheline per thread (global_prefetch_b8)
    if (j + 1 < NJ) __builtin_prefetch(wt + 32 * DDIM + tid * 32, 0, 1);
    __syncthreads();

    // ---- 16x16 tile product over D=256: 8 chunks x 3 split-WMMAs ----
    const unsigned short* bhb = whi_us + (c * 16 + r16) * 256;
    const unsigned short* blb = wlo_us + (c * 16 + r16) * 256;
    v8f acc = {};
    #pragma unroll
    for (int kc = 0; kc < 8; ++kc) {
      int g1 = ((4 * kc + kh) ^ r16) * 8;       // swizzled granule offsets
      int g2 = ((4 * kc + kh + 2) ^ r16) * 8;
      v16bf bh = ld2(bhb + g1, bhb + g2);
      v16bf bl = ld2(blb + g1, blb + g2);
      acc = __builtin_amdgcn_wmma_f32_16x16x32_bf16(false, ahi[kc], false, bh,
                                                    (short)0, acc, false, false);
      acc = __builtin_amdgcn_wmma_f32_16x16x32_bf16(false, ahi[kc], false, bl,
                                                    (short)0, acc, false, false);
      acc = __builtin_amdgcn_wmma_f32_16x16x32_bf16(false, alo[kc], false, bh,
                                                    (short)0, acc, false, false);
    }

    // score = ||e||^2 - 2 z.e (constant ||z||^2 dropped for argmin accuracy)
    const int   colg = (2 * j + c) * 16 + r16;
    const float wn   = wnorm[colg];
    #pragma unroll
    for (int r = 0; r < 8; ++r) {
      float d = wn - 2.0f * acc[r];
      if (d < best[r]) { best[r] = d; bidx[r] = colg; }
    }
  }

  // ---- cross-lane argmin within 16-lane halves (wave32), first-index ties
  #pragma unroll
  for (int m = 8; m >= 1; m >>= 1) {
    #pragma unroll
    for (int r = 0; r < 8; ++r) {
      float ov = __shfl_xor(best[r], m, 32);
      int   oi = __shfl_xor(bidx[r], m, 32);
      if (ov < best[r] || (ov == best[r] && oi < bidx[r])) {
        best[r] = ov; bidx[r] = oi;
      }
    }
  }

  // ---- cross-parity merge through LDS (reuse weight-tile region) ----
  __syncthreads();
  float* vbuf = (float*)smem_u;            // 2 x 64 floats
  int*   ibuf = (int*)(smem_u + 128);      // 2 x 64 ints
  if (r16 == 0) {
    // C layout: VGPR r -> row r (lanes 0-15) / r+8 (lanes 16-31)
    #pragma unroll
    for (int r = 0; r < 8; ++r) {
      int row_local = rw * 16 + kh * 8 + r;
      vbuf[c * 64 + row_local] = best[r];
      ibuf[c * 64 + row_local] = bidx[r];
    }
  }
  __syncthreads();
  if (tid < ROWS_PB) {
    float v0 = vbuf[tid], v1 = vbuf[64 + tid];
    int   i0 = ibuf[tid], i1 = ibuf[64 + tid];
    int win = (v1 < v0 || (v1 == v0 && i1 < i0)) ? i1 : i0;
    idx_out[row0 + tid]   = win;
    idx_f_out[row0 + tid] = (float)win;
  }
}

// ---------------- gather + loss partial + segment-sum scatter --------------
__global__ __launch_bounds__(256) void gather_kernel(
    const float* __restrict__ z, const float* __restrict__ w,
    const int* __restrict__ idx, float* __restrict__ q_out,
    float* __restrict__ dw, float* __restrict__ counts,
    float* __restrict__ loss_acc) {
  const int row = blockIdx.x;
  const int t   = threadIdx.x;
  const int id  = idx[row];
  float zv = z[(size_t)row * DDIM + t];
  float qv = w[(size_t)id * DDIM + t];
  q_out[(size_t)row * DDIM + t] = qv;  // quantized_st == quantized numerically

  float diff = qv - zv;
  float s = diff * diff;
  #pragma unroll
  for (int m = 16; m >= 1; m >>= 1) s += __shfl_xor(s, m, 32);
  __shared__ float sb[8];
  if ((t & 31) == 0) sb[t >> 5] = s;
  __syncthreads();
  if (t < 8) {
    float v = sb[t];
    #pragma unroll
    for (int m = 4; m >= 1; m >>= 1) v += __shfl_xor(v, m, 32);
    if (t == 0) atomicAdd(loss_acc, v);
  }
  atomicAdd(&dw[(size_t)id * DDIM + t], zv);
  if (t == 0) atomicAdd(&counts[id], 1.0f);
}

// -------- new_cluster_size + n = sum(new_cluster_size) ---------------------
__global__ __launch_bounds__(256) void cluster_kernel(
    const float* __restrict__ ema_cs, const float* __restrict__ counts,
    float* __restrict__ ncs_out, float* __restrict__ n_acc) {
  int k = blockIdx.x * blockDim.x + threadIdx.x;
  float v = 0.f;
  if (k < KCODES) {
    v = 0.99f * ema_cs[k] + 0.01f * counts[k];
    ncs_out[k] = v;
  }
  #pragma unroll
  for (int m = 16; m >= 1; m >>= 1) v += __shfl_xor(v, m, 32);
  __shared__ float sb[8];
  if ((threadIdx.x & 31) == 0) sb[threadIdx.x >> 5] = v;
  __syncthreads();
  if (threadIdx.x < 8) {
    float x = sb[threadIdx.x];
    #pragma unroll
    for (int m = 4; m >= 1; m >>= 1) x += __shfl_xor(x, m, 32);
    if (threadIdx.x == 0) atomicAdd(n_acc, x);
  }
}

// ---------------------------- new_ema_w ------------------------------------
__global__ __launch_bounds__(256) void emaw_kernel(
    const float* __restrict__ ema_w, const float* __restrict__ dw,
    float* __restrict__ new_ema_w) {
  size_t i = (size_t)blockIdx.x * 256 + threadIdx.x;
  if (i < (size_t)KCODES * DDIM) new_ema_w[i] = 0.99f * ema_w[i] + 0.01f * dw[i];
}

// -------------------- new_weight + scalar loss -----------------------------
__global__ __launch_bounds__(256) void finalize_kernel(
    const float* __restrict__ new_ema_w, const float* __restrict__ ncs,
    const float* __restrict__ n_acc, const float* __restrict__ loss_acc,
    float* __restrict__ new_weight, float* __restrict__ loss_out) {
  size_t i = (size_t)blockIdx.x * 256 + threadIdx.x;
  if (i < (size_t)KCODES * DDIM) {
    int   k  = (int)(i >> 8);
    float n  = *n_acc;
    float cs = (ncs[k] + 1e-5f) / (n + (float)KCODES * 1e-5f) * n;
    new_weight[i] = new_ema_w[i] / cs;
  }
  if (i == 0) *loss_out = 1.25f * (*loss_acc) / (float)((size_t)NROWS * DDIM);
}

// ---------------------------------------------------------------------------
extern "C" void kernel_launch(void* const* d_in, const int* in_sizes, int n_in,
                              void* d_out, int out_size, void* d_ws, size_t ws_size,
                              hipStream_t stream) {
  const float* z      = (const float*)d_in[0];  // [16,1024,256]
  const float* w      = (const float*)d_in[1];  // [8192,256]
  const float* ema_cs = (const float*)d_in[2];  // [8192]
  const float* ema_w  = (const float*)d_in[3];  // [8192,256]

  // outputs, concatenated in reference return order (all as f32)
  float* out    = (float*)d_out;
  float* o_q    = out;                               // 16*1024*256
  float* o_loss = o_q + (size_t)NROWS * DDIM;        // 1
  float* o_idx  = o_loss + 1;                        // 16384 (indices as f32)
  float* o_w    = o_idx + NROWS;                     // 8192*256
  float* o_cs   = o_w + (size_t)KCODES * DDIM;       // 8192
  float* o_emaw = o_cs + KCODES;                     // 8192*256

  // workspace
  float* ws       = (float*)d_ws;
  int*   s_idx    = (int*)ws;                        // 16384 ints
  float* s_wnorm  = ws + NROWS;                      // 8192
  float* s_counts = s_wnorm + KCODES;                // 8192
  float* s_dw     = s_counts + KCODES;               // 8192*256
  float* s_loss   = s_dw + (size_t)KCODES * DDIM;    // 1
  float* s_n      = s_loss + 1;                      // 1

  // 1) ||e||^2 per code
  wnorm_kernel<<<KCODES / 8, 256, 0, stream>>>(w, s_wnorm);

  // 2) zero accumulators (counts, dw, loss, n) — ws is poisoned by harness
  size_t nzero = (size_t)KCODES + (size_t)KCODES * DDIM + 2;
  zero_kernel<<<2048, 256, 0, stream>>>(s_counts, nzero);

  // 3) fused distance GEMM (bf16x3 WMMA) + argmin
  vq_argmin_kernel<<<NROWS / ROWS_PB, 256, 0, stream>>>(z, w, s_wnorm, s_idx, o_idx);

  // 4) gather quantized, loss partials, segment sums (counts, dw)
  gather_kernel<<<NROWS, 256, 0, stream>>>(z, w, s_idx, o_q, s_dw, s_counts, s_loss);

  // 5) EMA cluster sizes + n
  cluster_kernel<<<KCODES / 256, 256, 0, stream>>>(ema_cs, s_counts, o_cs, s_n);

  // 6) EMA codebook accumulator
  emaw_kernel<<<KCODES * DDIM / 256, 256, 0, stream>>>(ema_w, s_dw, o_emaw);

  // 7) normalized new codebook + scalar loss
  finalize_kernel<<<KCODES * DDIM / 256, 256, 0, stream>>>(o_emaw, o_cs, s_n,
                                                           s_loss, o_w, o_loss);
}
